// RNNClassifier_81733227643137
// MI455X (gfx1250) — matrix operand order repaired and checked by reading, not compiled, not run on hardware
//
#include <hip/hip_runtime.h>

typedef __attribute__((ext_vector_type(16))) __bf16 v16bf;
typedef __attribute__((ext_vector_type(8)))  __bf16 v8bf;
typedef __attribute__((ext_vector_type(8)))  float  v8f;
typedef __attribute__((ext_vector_type(4)))  float  v4f;

#define SEQ_T   512
#define BATCH_B 256
#define HID_H   1024
#define IN_I    256
#define OUT_O   2
#define NB_L1   64           // layer-1 workgroups, 16 columns each
#define NB_L2   64           // layer-2 workgroups, 16 columns each
#define NBLK    (NB_L1 + NB_L2)
#define NTHR    256          // 8 waves (wave32)
#define KH_IT   32           // 1024/32  (recurrent K part, both layers)
#define KX_IT   8            // 256/32   (x K part, layer 1)
#define K1IT    (KH_IT + KX_IT)
#define K2IT    64           // 2048/32  (layer 2 total)

// ---------------------------------------------------------------------------
// Device-wide sense barrier (persistent-kernel scan). Counter+generation in
// global workspace, zeroed by hipMemsetAsync before launch. s_sleep in spin.
// ---------------------------------------------------------------------------
__device__ __forceinline__ void grid_barrier(int* cnt, int* gen) {
  __syncthreads();
  if (threadIdx.x == 0) {
    __threadfence();  // make this WG's h stores visible device-wide
    int g = __hip_atomic_load(gen, __ATOMIC_RELAXED, __HIP_MEMORY_SCOPE_AGENT);
    int a = __hip_atomic_fetch_add(cnt, 1, __ATOMIC_ACQ_REL, __HIP_MEMORY_SCOPE_AGENT);
    if (a == NBLK - 1) {
      __hip_atomic_store(cnt, 0, __ATOMIC_RELAXED, __HIP_MEMORY_SCOPE_AGENT);
      __hip_atomic_fetch_add(gen, 1, __ATOMIC_RELEASE, __HIP_MEMORY_SCOPE_AGENT);
    } else {
      while (__hip_atomic_load(gen, __ATOMIC_ACQUIRE, __HIP_MEMORY_SCOPE_AGENT) == g)
        __builtin_amdgcn_s_sleep(2);
    }
  }
  __syncthreads();
  __threadfence();  // acquire: invalidate stale near-cache lines for all lanes
}

// A-operand gather, 16-bit A 16x32 layout (ISA 7.12.2):
// lanes<16: K 0..7 in v0..3, 16..23 in v4..7 ; lanes>=16: K 8..15, 24..31.
__device__ __forceinline__ v16bf load_a_bf16(const __bf16* rowp, int kb, int koff) {
  v8bf lo = *(const v8bf*)(rowp + kb + koff);
  v8bf hi = *(const v8bf*)(rowp + kb + 16 + koff);
  v16bf a;
#pragma unroll
  for (int i = 0; i < 8; ++i) { a[i] = lo[i]; a[i + 8] = hi[i]; }
  return a;
}

__device__ __forceinline__ v16bf load_a_f32(const float* rowp, int koff) {
  v4f f0 = *(const v4f*)(rowp + koff);
  v4f f1 = *(const v4f*)(rowp + koff + 4);
  v4f f2 = *(const v4f*)(rowp + 16 + koff);
  v4f f3 = *(const v4f*)(rowp + 16 + koff + 4);
  v16bf a;
#pragma unroll
  for (int i = 0; i < 4; ++i) {
    a[i]      = (__bf16)f0[i];
    a[i + 4]  = (__bf16)f1[i];
    a[i + 8]  = (__bf16)f2[i];
    a[i + 12] = (__bf16)f3[i];
  }
  return a;
}

__device__ __forceinline__ float sigmoidf_(float v) {
  return 1.0f / (1.0f + __expf(-v));
}

// One K-block step: B tile from LDS (shared by both row tiles), two WMMAs.
__device__ __forceinline__ void wmma_step_bf16(const __bf16* wlds, int i,
                                               int lane, int koff,
                                               const __bf16* pa0, const __bf16* pa1,
                                               int kb, v8f& acc0, v8f& acc1) {
  v16bf bm = *(const v16bf*)(wlds + i * 512 + lane * 16);
  v16bf a0 = load_a_bf16(pa0, kb, koff);
  v16bf a1 = load_a_bf16(pa1, kb, koff);
  acc0 = __builtin_amdgcn_wmma_f32_16x16x32_bf16(false, a0, false, bm,
                                                 (short)0, acc0, false, false);
  acc1 = __builtin_amdgcn_wmma_f32_16x16x32_bf16(false, a1, false, bm,
                                                 (short)0, acc1, false, false);
}

// ---------------------------------------------------------------------------
// Persistent scan kernel. Blocks [0,64): layer-1 columns; [64,128): layer-2.
// Pipelined: at iteration j, layer-1 computes h1 for step j while layer-2
// computes h2 for step j-1 (disjoint parity buffers) -> 1 barrier per step.
// ---------------------------------------------------------------------------
__global__ void __launch_bounds__(NTHR) rnn_scan(
    const float* __restrict__ x,
    const float* __restrict__ whh1W, const float* __restrict__ whh1b,
    const float* __restrict__ wxh1W, const float* __restrict__ wxh1b,
    const float* __restrict__ whh2W, const float* __restrict__ whh2b,
    const float* __restrict__ wxh2W, const float* __restrict__ wxh2b,
    __bf16* __restrict__ h1buf,   // [2][BATCH_B*HID_H]
    __bf16* __restrict__ h2buf,   // [2][BATCH_B*HID_H]
    int* __restrict__ counters)
{
  // Weight slice in bf16, pre-swizzled to WMMA-B lane layout: [kIter][lane][16]
  // lanes<16 hold column (lane&15) K kb+0..7,16..23; lanes>=16 hold K kb+8..15,24..31.
  __shared__ __bf16 wlds[K2IT * 32 * 16];   // 64 KB (layer2); layer1 uses 40 KB

  const bool isL2 = (blockIdx.x >= NB_L1);
  const int n0 = (isL2 ? (blockIdx.x - NB_L1) : blockIdx.x) * 16;
  const int tid = threadIdx.x;
  const int nIt = isL2 ? K2IT : K1IT;

  {  // one-time weight slice load: global f32 -> LDS bf16 (reused 512 steps)
    const float* Wh = isL2 ? whh2W : whh1W;       // [HID_H][HID_H]
    const float* Wx = isL2 ? wxh2W : wxh1W;       // [HID_H][HID_H or IN_I]
    const int xK = isL2 ? HID_H : IN_I;
    for (int e = tid; e < nIt * 32 * 16; e += NTHR) {
      int h = e & 15, l = (e >> 4) & 31, i = e >> 9;
      int k = i * 32 + ((h & 8) ? 16 : 0) + ((l & 16) ? 8 : 0) + (h & 7);
      int col = n0 + (l & 15);
      float v = (k < HID_H) ? Wh[(size_t)col * HID_H + k]
                            : Wx[(size_t)col * xK + (k - HID_H)];
      wlds[e] = (__bf16)v;
    }
  }

  const int lane = tid & 31;
  const int wave = tid >> 5;
  const int koff = (lane & 16) ? 8 : 0;   // A K-offset per lane half
  const int ccol = lane & 15;             // output column (C/D layout)
  const int rhi  = (lane & 16) ? 8 : 0;   // C/D row offset for upper lanes
  const float bias = isL2 ? (whh2b[n0 + ccol] + wxh2b[n0 + ccol])
                          : (whh1b[n0 + ccol] + wxh1b[n0 + ccol]);
  __syncthreads();

  const size_t HB = (size_t)BATCH_B * HID_H;
  int* cnt = counters;
  int* gen = counters + 1;
  const int r0 = wave * 32, r1 = r0 + 16;
  const size_t arow0 = (size_t)(r0 + ccol) * HID_H;
  const size_t arow1 = (size_t)(r1 + ccol) * HID_H;

  for (int j = 0; j <= SEQ_T; ++j) {
    const int p = j & 1;
    if (!isL2) {
      if (j < SEQ_T) {
        // h1_next = sigmoid([h1 | x_t] @ [Whh1 ; Wxh1]^T + b), K = 1280
        const __bf16* h1c = h1buf + (size_t)p * HB;
        __bf16*       h1n = h1buf + (size_t)(p ^ 1) * HB;
        const float*  xt  = x + (size_t)j * BATCH_B * IN_I;
        const __bf16* pa0 = h1c + arow0;
        const __bf16* pa1 = h1c + arow1;
        const float*  px0 = xt + (size_t)(r0 + ccol) * IN_I;
        const float*  px1 = xt + (size_t)(r1 + ccol) * IN_I;
        v8f acc0 = {}, acc1 = {};
        // K 0..1023: recurrent h1 part (bf16), branch-free loop
#pragma unroll 2
        for (int i = 0; i < KH_IT; ++i)
          wmma_step_bf16(wlds, i, lane, koff, pa0, pa1, i * 32, acc0, acc1);
        // K 1024..1279: x_t part (f32 -> bf16 on the fly), branch-free loop
#pragma unroll 2
        for (int i = 0; i < KX_IT; ++i) {
          v16bf bm = *(const v16bf*)(wlds + (KH_IT + i) * 512 + lane * 16);
          v16bf a0 = load_a_f32(px0 + i * 32, koff);
          v16bf a1 = load_a_f32(px1 + i * 32, koff);
          acc0 = __builtin_amdgcn_wmma_f32_16x16x32_bf16(false, a0, false, bm,
                                                         (short)0, acc0, false, false);
          acc1 = __builtin_amdgcn_wmma_f32_16x16x32_bf16(false, a1, false, bm,
                                                         (short)0, acc1, false, false);
        }
#pragma unroll
        for (int r = 0; r < 8; ++r) {
          float v0 = sigmoidf_(acc0[r] + bias);
          float v1 = sigmoidf_(acc1[r] + bias);
          h1n[(size_t)(r0 + rhi + r) * HID_H + n0 + ccol] = (__bf16)v0;
          h1n[(size_t)(r1 + rhi + r) * HID_H + n0 + ccol] = (__bf16)v1;
        }
      }
    } else {
      if (j >= 1) {
        // step t=j-1: h2_next = sigmoid([h2 | h1_new] @ [Whh2 ; Wxh2]^T + b), K = 2048
        const __bf16* h2c = h2buf + (size_t)(p ^ 1) * HB;
        __bf16*       h2n = h2buf + (size_t)p * HB;
        const __bf16* h1r = h1buf + (size_t)p * HB;   // h1 written at iter j-1
        const __bf16* pa0 = h2c + arow0;
        const __bf16* pa1 = h2c + arow1;
        const __bf16* pb0 = h1r + arow0;
        const __bf16* pb1 = h1r + arow1;
        v8f acc0 = {}, acc1 = {};
        // K 0..1023: recurrent h2 part
#pragma unroll 2
        for (int i = 0; i < KH_IT; ++i)
          wmma_step_bf16(wlds, i, lane, koff, pa0, pa1, i * 32, acc0, acc1);
        // K 1024..2047: fresh h1 part
#pragma unroll 2
        for (int i = 0; i < KH_IT; ++i)
          wmma_step_bf16(wlds, KH_IT + i, lane, koff, pb0, pb1, i * 32, acc0, acc1);
#pragma unroll
        for (int r = 0; r < 8; ++r) {
          float v0 = sigmoidf_(acc0[r] + bias);
          float v1 = sigmoidf_(acc1[r] + bias);
          h2n[(size_t)(r0 + rhi + r) * HID_H + n0 + ccol] = (__bf16)v0;
          h2n[(size_t)(r1 + rhi + r) * HID_H + n0 + ccol] = (__bf16)v1;
        }
      }
    }
    grid_barrier(cnt, gen);   // one device-wide barrier per pipelined step
  }
}

// hidden_state (f32, zeros) -> bf16 parity-0 state buffers
__global__ void init_h(const float* __restrict__ h0,
                       __bf16* __restrict__ h1buf, __bf16* __restrict__ h2buf) {
  int i = blockIdx.x * blockDim.x + threadIdx.x;
  float v = h0[i];
  h1buf[i] = (__bf16)v;
  h2buf[i] = (__bf16)v;
}

// logits = h2_final @ fc_W^T + fc_b ; softmax over the 2 classes
__global__ void fc_softmax(const __bf16* __restrict__ h2final,
                           const float* __restrict__ fcW,
                           const float* __restrict__ fcb,
                           float* __restrict__ out) {
  int b = threadIdx.x;  // 256 threads, one batch row each
  float a0 = fcb[0], a1 = fcb[1];
  const __bf16* hp = h2final + (size_t)b * HID_H;
  for (int k = 0; k < HID_H; ++k) {
    float hv = (float)hp[k];
    a0 += hv * fcW[k];
    a1 += hv * fcW[HID_H + k];
  }
  float m = fmaxf(a0, a1);
  float e0 = __expf(a0 - m), e1 = __expf(a1 - m);
  float inv = 1.0f / (e0 + e1);
  out[b * 2 + 0] = e0 * inv;
  out[b * 2 + 1] = e1 * inv;
}

extern "C" void kernel_launch(void* const* d_in, const int* in_sizes, int n_in,
                              void* d_out, int out_size, void* d_ws, size_t ws_size,
                              hipStream_t stream) {
  (void)in_sizes; (void)n_in; (void)out_size; (void)ws_size;
  const float* x     = (const float*)d_in[0];
  const float* h0    = (const float*)d_in[1];
  const float* whh1W = (const float*)d_in[2];
  const float* whh1b = (const float*)d_in[3];
  const float* wxh1W = (const float*)d_in[4];
  const float* wxh1b = (const float*)d_in[5];
  const float* whh2W = (const float*)d_in[6];
  const float* whh2b = (const float*)d_in[7];
  const float* wxh2W = (const float*)d_in[8];
  const float* wxh2b = (const float*)d_in[9];
  const float* fcW   = (const float*)d_in[10];
  const float* fcb   = (const float*)d_in[11];
  float* out = (float*)d_out;

  // Workspace: [0,256) barrier counters; then h1 ping-pong (1 MB); h2 (1 MB).
  char* ws = (char*)d_ws;
  int*    counters = (int*)ws;
  __bf16* h1buf = (__bf16*)(ws + 256);
  __bf16* h2buf = (__bf16*)(ws + 256 + 2 * sizeof(__bf16) * BATCH_B * HID_H);

  hipMemsetAsync(counters, 0, 256, stream);  // barrier state must start at 0
  init_h<<<dim3((BATCH_B * HID_H) / NTHR), dim3(NTHR), 0, stream>>>(h0, h1buf, h2buf);
  rnn_scan<<<dim3(NBLK), dim3(NTHR), 0, stream>>>(
      x, whh1W, whh1b, wxh1W, wxh1b, whh2W, whh2b, wxh2W, wxh2b,
      h1buf, h2buf, counters);
  // Final h2 lands in parity-0 buffer (t=511 written at iteration 512).
  fc_softmax<<<dim3(1), dim3(BATCH_B), 0, stream>>>(h2buf, fcW, fcb, out);
  // Second output of the reference tuple is the (unchanged) input hidden_state.
  hipMemcpyAsync(out + BATCH_B * OUT_O, h0, sizeof(float) * BATCH_B * HID_H,
                 hipMemcpyDeviceToDevice, stream);
}
